// MessageAggregation_26809185861893
// MI455X (gfx1250) — matrix operand order repaired and checked by reading, not compiled, not run on hardware
//
#include <hip/hip_runtime.h>

// ---------------------------------------------------------------------------
// Types for CDNA5 WMMA / TDM (gfx1250, wave32)
// ---------------------------------------------------------------------------
typedef __bf16  v16bf __attribute__((ext_vector_type(16)));
typedef __bf16  v8bf  __attribute__((ext_vector_type(8)));
typedef float   v8f   __attribute__((ext_vector_type(8)));
typedef unsigned int uint32x4 __attribute__((ext_vector_type(4)));
typedef int          int32x4  __attribute__((ext_vector_type(4)));
typedef int          int32x8  __attribute__((ext_vector_type(8)));

union BF16x16 { v16bf v; v8bf h[2]; unsigned short u[16]; };

__device__ __forceinline__ unsigned short f2bf(float f) {
  unsigned int u = __float_as_uint(f);
  u += 0x7FFFu + ((u >> 16) & 1u);           // round-to-nearest-even
  return (unsigned short)(u >> 16);
}

__device__ __forceinline__ v8f wmma_bf16(v16bf a, v16bf b, v8f c) {
  // D = A(16x32 bf16) * B(32x16 bf16) + C(16x16 f32)
  return __builtin_amdgcn_wmma_f32_16x16x32_bf16(false, a, false, b, (short)0, c,
                                                 false, false);
}

#define BATCH 16
#define NAG   128
#define DIM   256
#define PE    16
#define NHD   4
#define MDIM  32
#define HD    128
#define MI    272
#define G2    81
#define G2P   96
#define ROWS  (BATCH * NAG)   // 2048
#define RSCALE 0.17677669529663689f  // 1/sqrt(32)

// ---------------------------------------------------------------------------
// K0: convert weights to bf16, build positional K/V tables (81 -> pad 96 rows)
// ---------------------------------------------------------------------------
__global__ __launch_bounds__(256) void prep_kernel(
    const float* __restrict__ Wq,   // [128][256]
    const float* __restrict__ Wk,   // [128][272]
    const float* __restrict__ Wv,   // [128][272]
    const float* __restrict__ Wagg, // [128][128]
    const float* __restrict__ Wih,  // [768][128]
    const float* __restrict__ Whh,  // [768][256]
    const float* __restrict__ Wpos, // [16][81]
    const float* __restrict__ bpos, // [16]
    unsigned short* __restrict__ oWq, unsigned short* __restrict__ oWk,
    unsigned short* __restrict__ oWv, unsigned short* __restrict__ oWagg,
    unsigned short* __restrict__ oWih, unsigned short* __restrict__ oWhh,
    unsigned short* __restrict__ okpos,  // [96][128] bf16
    float* __restrict__ ovpos)           // [96][128] f32
{
  const int SQ = 128 * 256, SK = 128 * 256, SV = 128 * 256, SA = 128 * 128;
  const int SI = 768 * 128, SH = 768 * 256, ST = G2P * 128;
  const int TOT = SQ + SK + SV + SA + SI + SH + 2 * ST;
  int i = blockIdx.x * 256 + threadIdx.x;
  if (i >= TOT) return;
  if (i < SQ) { oWq[i] = f2bf(Wq[i]); return; }
  i -= SQ;
  if (i < SK) { oWk[i] = f2bf(Wk[(i >> 8) * MI + (i & 255)]); return; }
  i -= SK;
  if (i < SV) { oWv[i] = f2bf(Wv[(i >> 8) * MI + (i & 255)]); return; }
  i -= SV;
  if (i < SA) { oWagg[i] = f2bf(Wagg[i]); return; }
  i -= SA;
  if (i < SI) { oWih[i] = f2bf(Wih[i]); return; }
  i -= SI;
  if (i < SH) { oWhh[i] = f2bf(Whh[i]); return; }
  i -= SH;
  if (i < ST) {
    int e = i >> 7, c = i & 127;
    float s = 0.0f;
    if (e < G2) {
      #pragma unroll
      for (int p = 0; p < PE; ++p)
        s += (Wpos[p * G2 + e] + bpos[p]) * Wk[c * MI + DIM + p];
    }
    okpos[i] = f2bf(s);
    return;
  }
  i -= ST;
  {
    int e = i >> 7, c = i & 127;
    float s = 0.0f;
    if (e < G2) {
      #pragma unroll
      for (int p = 0; p < PE; ++p)
        s += (Wpos[p * G2 + e] + bpos[p]) * Wv[c * MI + DIM + p];
    }
    ovpos[i] = s;
  }
}

// ---------------------------------------------------------------------------
// K1: LayerNorm(hidden) -> x_bf16 ; hidden -> h_bf16
// ---------------------------------------------------------------------------
__global__ __launch_bounds__(256) void ln_kernel(
    const float* __restrict__ hidden, const float* __restrict__ lnw,
    const float* __restrict__ lnb, unsigned short* __restrict__ xb,
    unsigned short* __restrict__ hb)
{
  const int row = blockIdx.x;
  const int c = threadIdx.x;
  const float v = hidden[(size_t)row * DIM + c];
  __shared__ float r1[256];
  __shared__ float r2[256];
  r1[c] = v; r2[c] = v * v;
  __syncthreads();
  for (int s = 128; s > 0; s >>= 1) {
    if (c < s) { r1[c] += r1[c + s]; r2[c] += r2[c + s]; }
    __syncthreads();
  }
  const float mu = r1[0] * (1.0f / DIM);
  const float var = r2[0] * (1.0f / DIM) - mu * mu;
  const float rstd = rsqrtf(var + 1e-5f);
  const float x = (v - mu) * rstd * lnw[c] + lnb[c];
  xb[(size_t)row * DIM + c] = f2bf(x);
  hb[(size_t)row * DIM + c] = f2bf(v);
}

// ---------------------------------------------------------------------------
// K2: generic C[M,N] = A[M,K](bf16) * W[N,K](bf16)^T + bias ; one wave per
// 16x16 tile, K stepped by 32 through v_wmma_f32_16x16x32_bf16.
// ---------------------------------------------------------------------------
__global__ __launch_bounds__(256) void gemm_bf16_wmma(
    const unsigned short* __restrict__ A, const unsigned short* __restrict__ W,
    const float* __restrict__ bias, void* __restrict__ out,
    int M, int N, int K, int out_bf16)
{
  const int wave = threadIdx.x >> 5;
  const int lane = threadIdx.x & 31;
  const int tilesN = N >> 4;
  const int tile = blockIdx.x * 8 + wave;
  const int mt = tile / tilesN, nt = tile % tilesN;
  if (mt * 16 >= M) return;                    // wave-uniform
  const int m0 = mt * 16, n0 = nt * 16;
  const int mlane = lane & 15;
  const int hi = lane >> 4;
  const int koffA = hi * 16;                   // A: K 0-15 vs 16-31
  const int koffB = hi * 16;                   // B: K 0-15 vs 16-31
  const size_t arow = (size_t)(m0 + mlane) * K;
  const size_t brow = (size_t)(n0 + mlane) * K;

  v8f acc = {};
  for (int k0 = 0; k0 < K; k0 += 32) {
    BF16x16 a, b;
    a.h[0] = *(const v8bf*)(A + arow + k0 + koffA);
    a.h[1] = *(const v8bf*)(A + arow + k0 + koffA + 8);
    b.v    = *(const v16bf*)(W + brow + k0 + koffB);
    acc = wmma_bf16(a.v, b.v, acc);
  }
  const int ncol = n0 + mlane;
  const float bv = bias ? bias[ncol] : 0.0f;
  const int rbase = m0 + hi * 8;
  if (out_bf16) {
    unsigned short* o = (unsigned short*)out;
    #pragma unroll
    for (int r = 0; r < 8; ++r)
      o[(size_t)(rbase + r) * N + ncol] = f2bf(acc[r] + bv);
  } else {
    float* o = (float*)out;
    #pragma unroll
    for (int r = 0; r < 8; ++r)
      o[(size_t)(rbase + r) * N + ncol] = acc[r] + bv;
  }
}

// ---------------------------------------------------------------------------
// K3: fused attention per (batch, head). V tile staged into LDS via the
// Tensor Data Mover (tensor_load_to_lds + s_wait_tensorcnt). Per 32-receiver
// chunk: WMMA scores (Q x Kbase^T), WMMA positional scores (Q x kpos^T),
// masked softmax over 128 senders (4 threads per receiver, LDS combine),
// positional V gather accumulated with ds_add_f32 atomics,
// WMMA aggregation (W x Vbase) + positional part -> agg bf16.
// ---------------------------------------------------------------------------
__global__ __launch_bounds__(256) void attn_kernel(
    const unsigned short* __restrict__ qb,    // [2048][128] bf16
    const unsigned short* __restrict__ kb,    // [2048][128] bf16
    const unsigned short* __restrict__ vb,    // [2048][128] bf16
    const unsigned short* __restrict__ kpos,  // [96][128] bf16
    const float* __restrict__ vpos,           // [96][128] f32
    const int* __restrict__ relpos,           // [16][128][128][2]
    const unsigned char* __restrict__ mask,   // [16][128][128]
    unsigned short* __restrict__ aggout,      // [2048][128] bf16
    int* __restrict__ upd)                    // [2048]
{
  const int b = blockIdx.x, h = blockIdx.y;
  const int tid = threadIdx.x;
  const int wave = tid >> 5, lane = tid & 31;
  const int mlane = lane & 15;
  const int hi = lane >> 4;
  const int koffA = hi * 16;
  const int koffB = hi * 16;

  __shared__ float sS[32][132];                       // scores  [j][i]
  __shared__ float sQP[32][100];                      // q.kpos  [j][e]
  __shared__ __align__(32) unsigned short sW[32][128];  // weights [j][i] bf16
  __shared__ __align__(32) unsigned short sVT[32][128]; // V^T     [d][i] bf16
  __shared__ __align__(64) unsigned short sV[128][32];  // TDM-staged V tile
  __shared__ float aggP[32][32];                      // positional agg
  __shared__ float red1[32][4];
  __shared__ float red2[32][4];
  __shared__ int   updF[32];
  __shared__ unsigned char eIdx[32][128];

  // ---- Stage V_h (rows b*128..+127, cols h*32..+31 of [2048][128] bf16)
  // into LDS with one Tensor Data Mover descriptor (2D tile, data_size=2B).
  {
    const unsigned long long gaddr =
        (unsigned long long)(size_t)(vb + (size_t)(b * NAG) * HD + h * MDIM);
    const unsigned int lds_addr = (unsigned int)(size_t)(&sV[0][0]);
    const uint32x4 g0 = {
        1u,                                             // count=1, user desc
        lds_addr,                                       // LDS byte address
        (unsigned int)gaddr,                            // global_addr[31:0]
        ((unsigned int)(gaddr >> 32) & 0x01FFFFFFu) | 0x80000000u}; // type=2
    const int32x8 g1 = {
        (1 << 16),            // workgroup_mask=0, data_size=1 (2 bytes)
        (HD << 16),           // tensor_dim0 = 128 (lo16 at bit 48)
        (ROWS << 16),         // tensor_dim0 hi=0, tensor_dim1 = 2048 (lo16)
        (MDIM << 16),         // tensor_dim1 hi=0, tile_dim0 = 32
        NAG,                  // tile_dim1 = 128, tile_dim2 = 0
        HD,                   // tensor_dim0_stride = 128
        0, 0};                // stride hi, tensor_dim1_stride (unused, 2D)
    const int32x4 gz = {0, 0, 0, 0};
    if (wave == 0) {                          // wave-uniform, one DMA issue
#if defined(__clang_major__) && __clang_major__ >= 23
      const int32x8 gz8 = {0, 0, 0, 0, 0, 0, 0, 0};
      __builtin_amdgcn_tensor_load_to_lds(g0, g1, gz, gz, gz8, 0);
#else
      __builtin_amdgcn_tensor_load_to_lds(g0, g1, gz, gz, 0);
#endif
      __builtin_amdgcn_s_wait_tensorcnt(0);
    }
  }
  __syncthreads();
  // transpose staged tile: sVT[d][i] = sV[i][d]  (LDS -> LDS)
  for (int t = tid; t < 32 * 128; t += 256) {
    int d = t >> 7, i = t & 127;
    sVT[d][i] = sV[i][d];
  }
  __syncthreads();

  for (int j0 = 0; j0 < NAG; j0 += 32) {
    for (int t = tid; t < 32 * 32; t += 256) aggP[t >> 5][t & 31] = 0.0f;
    if (tid < 32) updF[tid] = 0;

    // ---- Phase A: S_base[j,i] = q[b,j0+j,h] . k_base[b,i,h]   (16 tiles)
    #pragma unroll
    for (int tt = 0; tt < 2; ++tt) {
      const int tile = wave * 2 + tt;          // 0..15
      const int m0 = (tile >> 3) * 16;         // j tile
      const int n0 = (tile & 7) * 16;          // i tile
      BF16x16 a, bm;
      const size_t arow = (size_t)(b * NAG + j0 + m0 + mlane) * HD + h * MDIM;
      a.h[0] = *(const v8bf*)(qb + arow + koffA);
      a.h[1] = *(const v8bf*)(qb + arow + koffA + 8);
      bm.v = *(const v16bf*)(kb + (size_t)(b * NAG + n0 + mlane) * HD +
                             h * MDIM + koffB);
      v8f c = {};
      c = wmma_bf16(a.v, bm.v, c);
      #pragma unroll
      for (int r = 0; r < 8; ++r) sS[m0 + hi * 8 + r][n0 + mlane] = c[r];
    }
    // ---- Phase B: QP[j,e] = q[b,j0+j,h] . kpos[e,h]   (12 tiles)
    #pragma unroll
    for (int tt = 0; tt < 2; ++tt) {
      const int tile = wave + tt * 8;
      if (tile < 12) {                         // wave-uniform
        const int jt = (tile >= 6) ? 1 : 0;
        const int et = tile - jt * 6;
        const int m0 = jt * 16, e0 = et * 16;
        BF16x16 a, bm;
        const size_t arow = (size_t)(b * NAG + j0 + m0 + mlane) * HD + h * MDIM;
        a.h[0] = *(const v8bf*)(qb + arow + koffA);
        a.h[1] = *(const v8bf*)(qb + arow + koffA + 8);
        bm.v = *(const v16bf*)(kpos + (size_t)(e0 + mlane) * HD +
                               h * MDIM + koffB);
        v8f c = {};
        c = wmma_bf16(a.v, bm.v, c);
        #pragma unroll
        for (int r = 0; r < 8; ++r) sQP[m0 + hi * 8 + r][e0 + mlane] = c[r];
      }
    }
    __syncthreads();

    // ---- Phase C: masked softmax over senders i (4 threads per receiver j)
    const bool act = tid < 128;
    const int jj = tid >> 2, q4 = tid & 3;
    const int jg = j0 + jj;
    float mj = -3.0e38f;
    if (act) {
      float mx = -3.0e38f;
      int anyv = 0;
      for (int i = q4 * 32; i < q4 * 32 + 32; ++i) {
        const size_t pidx = (size_t)(b * NAG + i) * NAG + jg;
        const int valid = mask[pidx] != 0;
        anyv |= valid;
        const int r0 = relpos[pidx * 2], r1 = relpos[pidx * 2 + 1];
        const int oob = (r0 < -4) | (r0 > 4) | (r1 < -4) | (r1 > 4);
        const int e = oob ? 40 : (r0 + 4) + (r1 + 4) * 9;
        eIdx[jj][i] = (unsigned char)e;
        if (valid)  // warm L2/L0 for the pass-3 positional V gather
          __builtin_prefetch(vpos + (size_t)e * HD + h * MDIM, 0, 1);
        float s = (sS[jj][i] + sQP[jj][e]) * RSCALE;
        s = valid ? s : -3.0e38f;
        sS[jj][i] = s;
        mx = fmaxf(mx, s);
      }
      red1[jj][q4] = mx;
      if (anyv) atomicOr(&updF[jj], 1);
    }
    __syncthreads();
    if (act) {
      mj = fmaxf(fmaxf(red1[jj][0], red1[jj][1]),
                 fmaxf(red1[jj][2], red1[jj][3]));
      float psum = 0.0f;
      if (mj > -1.0e37f) {
        for (int i = q4 * 32; i < q4 * 32 + 32; ++i) {
          const float s = sS[jj][i];
          psum += (s > -1.0e37f) ? __expf(s - mj) : 0.0f;
        }
      }
      red2[jj][q4] = psum;
    }
    __syncthreads();
    if (act) {
      const float sum = red2[jj][0] + red2[jj][1] + red2[jj][2] + red2[jj][3];
      const float inv = (sum > 0.0f) ? 1.0f / sum : 0.0f;
      float accv[32];
      #pragma unroll
      for (int d = 0; d < 32; ++d) accv[d] = 0.0f;
      for (int i = q4 * 32; i < q4 * 32 + 32; ++i) {
        const float s = sS[jj][i];
        const float w =
            (mj > -1.0e37f && s > -1.0e37f) ? __expf(s - mj) * inv : 0.0f;
        sW[jj][i] = f2bf(w);
        if (w != 0.0f) {
          const float4* vp =
              (const float4*)(vpos + (size_t)eIdx[jj][i] * HD + h * MDIM);
          #pragma unroll
          for (int d4 = 0; d4 < 8; ++d4) {
            const float4 vv = vp[d4];
            accv[d4 * 4 + 0] += w * vv.x;
            accv[d4 * 4 + 1] += w * vv.y;
            accv[d4 * 4 + 2] += w * vv.z;
            accv[d4 * 4 + 3] += w * vv.w;
          }
        }
      }
      #pragma unroll
      for (int d = 0; d < 32; ++d) atomicAdd(&aggP[jj][d], accv[d]);
    }
    __syncthreads();

    // ---- Phase D: agg[j,d] = sum_i W[j,i] * Vbase[i,d]  (+ positional part)
    if (wave < 4) {                            // wave-uniform
      const int m0 = (wave >> 1) * 16;         // j tile
      const int n0 = (wave & 1) * 16;          // d tile
      v8f c = {};
      #pragma unroll
      for (int kk = 0; kk < 128; kk += 32) {
        BF16x16 a, bm;
        a.h[0] = *(const v8bf*)(&sW[m0 + mlane][kk + koffA]);
        a.h[1] = *(const v8bf*)(&sW[m0 + mlane][kk + koffA + 8]);
        bm.v   = *(const v16bf*)(&sVT[n0 + mlane][kk + koffB]);
        c = wmma_bf16(a.v, bm.v, c);
      }
      #pragma unroll
      for (int r = 0; r < 8; ++r) {
        const int jl = m0 + hi * 8 + r;
        const int dl = n0 + mlane;
        const float val = c[r] + aggP[jl][dl];
        aggout[(size_t)(b * NAG + j0 + jl) * HD + h * MDIM + dl] = f2bf(val);
      }
    }
    if (h == 0 && tid < 32) upd[b * NAG + j0 + tid] = updF[tid];
    __syncthreads();
  }
}

// ---------------------------------------------------------------------------
// K5: GRU gates + update-mask select
// ---------------------------------------------------------------------------
__global__ __launch_bounds__(256) void gru_kernel(
    const float* __restrict__ gi, const float* __restrict__ gh,
    const float* __restrict__ hidden, const int* __restrict__ upd,
    float* __restrict__ out)
{
  const size_t i = (size_t)blockIdx.x * 256 + threadIdx.x;
  if (i >= (size_t)ROWS * DIM) return;
  const int r = (int)(i >> 8), c = (int)(i & 255);
  const size_t base = (size_t)r * 768;
  const float gir = gi[base + c], giz = gi[base + 256 + c],
              gin = gi[base + 512 + c];
  const float ghr = gh[base + c], ghz = gh[base + 256 + c],
              ghn = gh[base + 512 + c];
  const float rg = 1.0f / (1.0f + __expf(-(gir + ghr)));
  const float zg = 1.0f / (1.0f + __expf(-(giz + ghz)));
  const float ng = tanhf(gin + rg * ghn);
  const float hf = hidden[i];
  const float hn = (1.0f - zg) * ng + zg * hf;
  out[i] = upd[r] ? hn : hf;
}

// ---------------------------------------------------------------------------
// Host-side launch
// ---------------------------------------------------------------------------
extern "C" void kernel_launch(void* const* d_in, const int* in_sizes, int n_in,
                              void* d_out, int out_size, void* d_ws,
                              size_t ws_size, hipStream_t stream) {
  (void)in_sizes; (void)n_in; (void)out_size; (void)ws_size;
  const float* hidden   = (const float*)d_in[0];
  const int*   relpos   = (const int*)d_in[1];
  const unsigned char* cmask = (const unsigned char*)d_in[2];
  const float* ln_w  = (const float*)d_in[3];
  const float* ln_b  = (const float*)d_in[4];
  const float* W_pos = (const float*)d_in[5];
  const float* b_pos = (const float*)d_in[6];
  const float* W_k   = (const float*)d_in[7];
  const float* b_k   = (const float*)d_in[8];
  const float* W_v   = (const float*)d_in[9];
  const float* b_v   = (const float*)d_in[10];
  const float* W_q   = (const float*)d_in[11];
  const float* b_q   = (const float*)d_in[12];
  const float* W_agg = (const float*)d_in[13];
  const float* b_agg = (const float*)d_in[14];
  const float* W_ih  = (const float*)d_in[15];
  const float* b_ih  = (const float*)d_in[16];
  const float* W_hh  = (const float*)d_in[17];
  const float* b_hh  = (const float*)d_in[18];
  float* out = (float*)d_out;

  // workspace carve-up (256B aligned slices)
  char* w = (char*)d_ws;
  size_t off = 0;
  auto take = [&](size_t bytes) -> char* {
    char* p = w + off;
    off += (bytes + 255) & ~(size_t)255;
    return p;
  };
  unsigned short* x_bf   = (unsigned short*)take((size_t)ROWS * DIM * 2);
  unsigned short* h_bf   = (unsigned short*)take((size_t)ROWS * DIM * 2);
  unsigned short* q_bf   = (unsigned short*)take((size_t)ROWS * HD * 2);
  unsigned short* k_bf   = (unsigned short*)take((size_t)ROWS * HD * 2);
  unsigned short* v_bf   = (unsigned short*)take((size_t)ROWS * HD * 2);
  unsigned short* Wq_b   = (unsigned short*)take((size_t)HD * DIM * 2);
  unsigned short* Wk_b   = (unsigned short*)take((size_t)HD * DIM * 2);
  unsigned short* Wv_b   = (unsigned short*)take((size_t)HD * DIM * 2);
  unsigned short* Wagg_b = (unsigned short*)take((size_t)HD * HD * 2);
  unsigned short* Wih_b  = (unsigned short*)take((size_t)768 * HD * 2);
  unsigned short* Whh_b  = (unsigned short*)take((size_t)768 * DIM * 2);
  unsigned short* kpos_b = (unsigned short*)take((size_t)G2P * HD * 2);
  float*          vpos_f = (float*)take((size_t)G2P * HD * 4);
  unsigned short* agg_b  = (unsigned short*)take((size_t)ROWS * HD * 2);
  unsigned short* agg2_b = (unsigned short*)take((size_t)ROWS * HD * 2);
  float*          gi_f   = (float*)take((size_t)ROWS * 768 * 4);
  float*          gh_f   = (float*)take((size_t)ROWS * 768 * 4);
  int*            upd_i  = (int*)take((size_t)ROWS * 4);

  // K0: weight conversion + positional tables
  {
    const int TOT = 128 * 256 * 3 + 128 * 128 + 768 * 128 + 768 * 256 +
                    2 * G2P * 128;
    prep_kernel<<<(TOT + 255) / 256, 256, 0, stream>>>(
        W_q, W_k, W_v, W_agg, W_ih, W_hh, W_pos, b_pos, Wq_b, Wk_b, Wv_b,
        Wagg_b, Wih_b, Whh_b, kpos_b, vpos_f);
  }
  // K1: layernorm + bf16 casts
  ln_kernel<<<ROWS, 256, 0, stream>>>(hidden, ln_w, ln_b, x_bf, h_bf);

  // K2: q/k/v base projections  (M=2048, N=128, K=256)
  {
    const int tiles = (ROWS / 16) * (HD / 16);  // 1024
    gemm_bf16_wmma<<<tiles / 8, 256, 0, stream>>>(x_bf, Wq_b, b_q, q_bf,
                                                  ROWS, HD, DIM, 1);
    gemm_bf16_wmma<<<tiles / 8, 256, 0, stream>>>(x_bf, Wk_b, b_k, k_bf,
                                                  ROWS, HD, DIM, 1);
    gemm_bf16_wmma<<<tiles / 8, 256, 0, stream>>>(x_bf, Wv_b, b_v, v_bf,
                                                  ROWS, HD, DIM, 1);
  }

  // K3: fused attention (per batch x head)
  attn_kernel<<<dim3(BATCH, NHD), 256, 0, stream>>>(
      q_bf, k_bf, v_bf, kpos_b, vpos_f, relpos, cmask, agg_b, upd_i);

  // K2: agg @ W_agg^T + b_agg  (M=2048, N=128, K=128) -> bf16
  {
    const int tiles = (ROWS / 16) * (HD / 16);
    gemm_bf16_wmma<<<tiles / 8, 256, 0, stream>>>(agg_b, Wagg_b, b_agg,
                                                  agg2_b, ROWS, HD, HD, 1);
  }
  // K2: GRU input/hidden projections -> f32
  {
    const int tiles = (ROWS / 16) * (768 / 16);  // 6144
    gemm_bf16_wmma<<<tiles / 8, 256, 0, stream>>>(agg2_b, Wih_b, b_ih, gi_f,
                                                  ROWS, 768, HD, 0);
    gemm_bf16_wmma<<<tiles / 8, 256, 0, stream>>>(h_bf, Whh_b, b_hh, gh_f,
                                                  ROWS, 768, DIM, 0);
  }
  // K5: GRU gates + update mask
  gru_kernel<<<(ROWS * DIM) / 256, 256, 0, stream>>>(gi_f, gh_f, hidden,
                                                     upd_i, out);
}
